// Decoder_3255585210458
// MI455X (gfx1250) — matrix operand order repaired and checked by reading, not compiled, not run on hardware
//
#include <hip/hip_runtime.h>
#include <hip/hip_bf16.h>
#include <math.h>

// ---------------- constants (match reference) ----------------
constexpr int kB   = 128;
constexpr int kS   = 128;
constexpr int kL   = 31;
constexpr int kD   = 384;
constexpr int kH   = 8;
constexpr int kF   = 24;
constexpr int kNB  = 6;
constexpr int kMLP = 1536;
constexpr int kHF  = kH * kF;     // 192
constexpr int kBS  = kB * kS;     // 16384
constexpr int TOK_PAD = 59;
constexpr int TOK_SOS = 60;

typedef __bf16 bf16_t;
typedef __attribute__((ext_vector_type(16))) __bf16 v16bf;
typedef __attribute__((ext_vector_type(8)))  __bf16 v8bf;
typedef __attribute__((ext_vector_type(8)))  float  v8f;
typedef __attribute__((ext_vector_type(4))) unsigned int u32x4;
typedef __attribute__((ext_vector_type(8))) int i32x8;
typedef __attribute__((ext_vector_type(4))) int i32x4;

// ---------------- helpers ----------------
__device__ inline bf16_t f2bf(float f) {
  unsigned u = __float_as_uint(f);
  u += 0x7FFFu + ((u >> 16) & 1u);          // round-to-nearest-even
  unsigned short h = (unsigned short)(u >> 16);
  return __builtin_bit_cast(bf16_t, h);
}

// A fragment: 16x32 bf16 tile, rows = base + (lane&15); per ISA 7.12.2:
// half = lane>>4; VGPRs 0..3 hold K = half*8 + 0..7, VGPRs 4..7 hold K = 16 + half*8 + 0..7
__device__ inline v16bf frag_a(const bf16_t* __restrict__ p, int row_base, int ld, int k_base) {
  int lane = threadIdx.x & 31;
  int m = lane & 15;
  int h = lane >> 4;
  const bf16_t* r = p + (size_t)(row_base + m) * ld + k_base + h * 8;
  v8bf lo = *(const v8bf*)(r);
  v8bf hi = *(const v8bf*)(r + 16);
  v16bf f;
#pragma unroll
  for (int i = 0; i < 8; ++i) { f[i] = lo[i]; f[i + 8] = hi[i]; }
  return f;
}

// B fragment: 32x16 bf16 tile from Bt (row = n), same K order as frag_a:
// half = lane>>4; elems 0..7 hold K = half*8 + 0..7, elems 8..15 hold K = 16 + half*8 + 0..7, n = lane&15
__device__ inline v16bf frag_b(const bf16_t* __restrict__ p, int n_base, int ld, int k_base) {
  int lane = threadIdx.x & 31;
  int n = lane & 15;
  int h = lane >> 4;
  const bf16_t* r = p + (size_t)(n_base + n) * ld + k_base + h * 8;
  v8bf lo = *(const v8bf*)(r);
  v8bf hi = *(const v8bf*)(r + 16);
  v16bf f;
#pragma unroll
  for (int i = 0; i < 8; ++i) { f[i] = lo[i]; f[i + 8] = hi[i]; }
  return f;
}

__device__ inline v8f wmma_bf16(v16bf a, v16bf b, v8f c) {
  return __builtin_amdgcn_wmma_f32_16x16x32_bf16(false, a, false, b, (short)0, c,
                                                 false, false);
}

__device__ inline float gelu_exact(float x) {
  return 0.5f * x * (1.0f + erff(x * 0.70710678118654752f));
}

// ---------------- TDM 2D tile load: global -> LDS ----------------
// D# per ISA ch.8: group0 = {count=1, lds_addr, global_addr[56:0], type=2},
// group1 = {data_size=2B, tensor_dim0=K, tensor_dim1=rows, tile 32 x rows,
//           tensor_dim0_stride=K}. Remaining groups zero (2D tile, tile_dim2=0).
// Toolchain uses the 6-arg builtin form.
__device__ inline void tdm_load_2d(unsigned lds_byte_addr, const void* gptr,
                                   unsigned tile_w, unsigned tile_h,
                                   unsigned stride_elems) {
  unsigned long long ga = (unsigned long long)(size_t)gptr;
  u32x4 g0;
  g0[0] = 1u;                                             // count=1 (valid user D#)
  g0[1] = lds_byte_addr;                                  // lds_addr
  g0[2] = (unsigned)ga;                                   // global_addr[31:0]
  g0[3] = (unsigned)((ga >> 32) & 0x01FFFFFFull) | (2u << 30);  // addr[56:32], type=2
  unsigned td0 = stride_elems;                            // tensor_dim0 (row length)
  unsigned td1 = tile_h;                                  // tensor_dim1 (rows)
  i32x8 g1;
  g1[0] = (int)(1u << 16);                                // data_size=1 -> 2-byte elems
  g1[1] = (int)((td0 & 0xFFFFu) << 16);                   // tensor_dim0 lo16 (atomic addr 0)
  g1[2] = (int)((td0 >> 16) | ((td1 & 0xFFFFu) << 16));   // dim0 hi16 | dim1 lo16
  g1[3] = (int)((td1 >> 16) | (tile_w << 16));            // dim1 hi16 | tile_dim0
  g1[4] = (int)(tile_h & 0xFFFFu);                        // tile_dim1 (tile_dim2=0)
  g1[5] = (int)stride_elems;                              // tensor_dim0_stride[31:0]
  g1[6] = 0;                                              // stride[47:32]=0, dim1_stride lo=0
  g1[7] = 0;
  i32x4 z4 = {0, 0, 0, 0};
  i32x8 z8 = {0, 0, 0, 0, 0, 0, 0, 0};
  __builtin_amdgcn_tensor_load_to_lds(g0, g1, z4, z4, z8, 0);
}

// ---------------- embedding ----------------
__global__ __launch_bounds__(128) void embed_kernel(
    const int* __restrict__ phrase, const float* __restrict__ pos_emb,
    const float* __restrict__ char_emb, float* __restrict__ x,
    bf16_t* __restrict__ xb) {
  int row = blockIdx.x;            // b*S + s
  int b = row / kS, s = row - b * kS;
  int tok = (s == 0) ? TOK_SOS : ((s - 1) < kL ? phrase[b * kL + (s - 1)] : TOK_PAD);
  const float* pr = pos_emb + (size_t)s * kD;
  const float* cr = char_emb + (size_t)tok * kD;
  float* xr = x + (size_t)row * kD;
  bf16_t* xbr = xb + (size_t)row * kD;
#pragma unroll
  for (int i = 0; i < 3; ++i) {
    int d = threadIdx.x + i * 128;
    float v = pr[d] + cr[d];
    xr[d] = v;
    xbr[d] = f2bf(v);
  }
}

// ---------------- fp32 -> bf16 plain convert ----------------
__global__ __launch_bounds__(256) void cvt_kernel(const float* __restrict__ in,
                                                  bf16_t* __restrict__ out, int n) {
  int i = blockIdx.x * 256 + threadIdx.x;
  if (i < n) out[i] = f2bf(in[i]);
}

// ---------------- fp32 [batch][R][C] -> bf16 [batch][C][R] (Bt layout) ----------------
__global__ __launch_bounds__(256) void transpose_cvt_kernel(
    const float* __restrict__ in, bf16_t* __restrict__ out, int R, int C) {
  const float* ib = in + (size_t)blockIdx.y * R * C;
  bf16_t* ob = out + (size_t)blockIdx.y * R * C;
  int idx = blockIdx.x * 256 + threadIdx.x;
  if (idx < R * C) {
    int r = idx / C, c = idx - r * C;
    ob[(size_t)c * R + r] = f2bf(ib[idx]);
  }
}

// ---------------- WMMA GEMM: C[M,N] = A[M,K] * Bt[N,K]^T ----------------
// mode 0: fp32 out, mode 1: bf16 out, mode 2: GELU + bf16 out
// M multiple of 128, N multiple of 64, K multiple of 32.
// Tiles staged by the Tensor Data Mover (double-buffered), waves only do WMMA.
__global__ __launch_bounds__(128) void gemm_bf16_kernel(
    const bf16_t* __restrict__ A, const bf16_t* __restrict__ Bt,
    void* __restrict__ Cout, int M, int N, int K, int mode) {
  __shared__ bf16_t As[2][128 * 32];
  __shared__ bf16_t Bs[2][64 * 32];
  const int bm = blockIdx.y * 128;
  const int bn = blockIdx.x * 64;
  const int wave = threadIdx.x >> 5;
  const int wm = (wave >> 1) * 64;   // 2 waves along M (64 rows each)
  const int wn = (wave & 1) * 32;    // 2 waves along N (32 cols each)
  const bool issuer = (wave == 0);
  const int nsteps = K >> 5;

  v8f acc[4][2] = {};

  if (issuer) {
    tdm_load_2d((unsigned)(size_t)&As[0][0], A + (size_t)bm * K, 32, 128, (unsigned)K);
    tdm_load_2d((unsigned)(size_t)&Bs[0][0], Bt + (size_t)bn * K, 32, 64, (unsigned)K);
  }

  for (int s = 0; s < nsteps; ++s) {
    const int cur = s & 1;
    if (issuer) {
      if (s + 1 < nsteps) {
        const int k1 = (s + 1) << 5;
        tdm_load_2d((unsigned)(size_t)&As[cur ^ 1][0],
                    A + (size_t)bm * K + k1, 32, 128, (unsigned)K);
        tdm_load_2d((unsigned)(size_t)&Bs[cur ^ 1][0],
                    Bt + (size_t)bn * K + k1, 32, 64, (unsigned)K);
        __builtin_amdgcn_s_wait_tensorcnt(2);  // oldest 2 (current bufs) complete
      } else {
        __builtin_amdgcn_s_wait_tensorcnt(0);
      }
    }
    __syncthreads();

    v16bf b0 = frag_b(&Bs[cur][0], wn + 0, 32, 0);
    v16bf b1 = frag_b(&Bs[cur][0], wn + 16, 32, 0);
#pragma unroll
    for (int mi = 0; mi < 4; ++mi) {
      v16bf a = frag_a(&As[cur][0], wm + mi * 16, 32, 0);
      acc[mi][0] = wmma_bf16(a, b0, acc[mi][0]);
      acc[mi][1] = wmma_bf16(a, b1, acc[mi][1]);
    }
    __syncthreads();
  }

  const int lane = threadIdx.x & 31;
  const int half = lane >> 4;
  const int nn = lane & 15;
#pragma unroll
  for (int mi = 0; mi < 4; ++mi) {
#pragma unroll
    for (int ni = 0; ni < 2; ++ni) {
#pragma unroll
      for (int r = 0; r < 8; ++r) {
        int row = bm + wm + mi * 16 + r + half * 8;
        int col = bn + wn + ni * 16 + nn;
        float v = acc[mi][ni][r];
        if (mode == 0) {
          ((float*)Cout)[(size_t)row * N + col] = v;
        } else if (mode == 1) {
          ((bf16_t*)Cout)[(size_t)row * N + col] = f2bf(v);
        } else {
          ((bf16_t*)Cout)[(size_t)row * N + col] = f2bf(gelu_exact(v));
        }
      }
    }
  }
}

// ---------------- fused attention: one workgroup per (b, h) ----------------
// scores = Q K^T / sqrt(F) with causal mask, softmax, out = P V
__global__ __launch_bounds__(256) void attn_kernel(
    const bf16_t* __restrict__ Q,   // [BS, HF]
    const bf16_t* __restrict__ Kb,  // [BS, HF]
    const bf16_t* __restrict__ Vb,  // [BS, HF]
    bf16_t* __restrict__ Ob) {      // [BS, HF]
  __shared__ bf16_t Qs[128 * 32];   // [q_s][f] (F padded to 32)
  __shared__ bf16_t Ks[128 * 32];   // [k_s][f]
  __shared__ bf16_t Vt[32 * 128];   // [f][k_s] (transposed for Bt layout)
  __shared__ bf16_t Ps[128 * 128];  // softmax probs, bf16

  const int h = blockIdx.x;
  const int b = blockIdx.y;
  const int tid = threadIdx.x;
  const bf16_t bz = f2bf(0.0f);

  const bf16_t* qp = Q + (size_t)(b * kS) * kHF + h * kF;
  const bf16_t* kp = Kb + (size_t)(b * kS) * kHF + h * kF;
  const bf16_t* vp = Vb + (size_t)(b * kS) * kHF + h * kF;

  // load Q,K tiles (vectorized 8-wide; f chunks 0,8,16 valid, 24 zero)
  for (int idx = tid; idx < 128 * 4; idx += 256) {
    int row = idx >> 2;
    int c = (idx & 3) * 8;
    v8bf qv, kv;
    if (c < kF) {
      qv = *(const v8bf*)(qp + (size_t)row * kHF + c);
      kv = *(const v8bf*)(kp + (size_t)row * kHF + c);
    } else {
#pragma unroll
      for (int j = 0; j < 8; ++j) { qv[j] = bz; kv[j] = bz; }
    }
    *(v8bf*)(&Qs[row * 32 + c]) = qv;
    *(v8bf*)(&Ks[row * 32 + c]) = kv;
  }
  // load V transposed: Vt[f][s]
  for (int idx = tid; idx < 32 * 128; idx += 256) {
    int f = idx >> 7;
    int s = idx & 127;
    Vt[f * 128 + s] = (f < kF) ? vp[(size_t)s * kHF + f] : bz;
  }
  __syncthreads();

  const int wave = tid >> 5;       // 0..7, owns query rows 16w..16w+15
  const int lane = tid & 31;
  const int half = lane >> 4;
  const int nn = lane & 15;
  const float scale = 0.20412414523193154f;   // 1/sqrt(24)

  // scores for this wave's 16x128 slab (single K-step, F padded to 32)
  v16bf aq = frag_a(Qs, wave * 16, 32, 0);
  v8f sc[8];
#pragma unroll
  for (int tn = 0; tn < 8; ++tn) {
    v8f z = {};
    sc[tn] = wmma_bf16(aq, frag_b(Ks, tn * 16, 32, 0), z);
  }

  // mask + softmax per row (row = wave*16 + r + 8*half; cols = tn*16 + nn)
#pragma unroll
  for (int r = 0; r < 8; ++r) {
    int row = wave * 16 + r + half * 8;
    float vals[8];
    float mx = -3.0e38f;
#pragma unroll
    for (int tn = 0; tn < 8; ++tn) {
      int col = tn * 16 + nn;
      float v = sc[tn][r] * scale;
      if (col > row) v = -1.0e9f;   // causal mask
      vals[tn] = v;
      mx = fmaxf(mx, v);
    }
#pragma unroll
    for (int m = 1; m < 16; m <<= 1) mx = fmaxf(mx, __shfl_xor(mx, m, 32));
    float sum = 0.0f;
#pragma unroll
    for (int tn = 0; tn < 8; ++tn) {
      float e = __expf(vals[tn] - mx);
      vals[tn] = e;
      sum += e;
    }
#pragma unroll
    for (int m = 1; m < 16; m <<= 1) sum += __shfl_xor(sum, m, 32);
    float inv = 1.0f / sum;
#pragma unroll
    for (int tn = 0; tn < 8; ++tn)
      Ps[row * 128 + tn * 16 + nn] = f2bf(vals[tn] * inv);
  }
  __syncthreads();

  // out = P V : this wave's 16 rows x 32 cols (f 0..31, store f<24)
  v8f o[2] = {};
#pragma unroll
  for (int ks = 0; ks < 4; ++ks) {
    v16bf ap = frag_a(Ps, wave * 16, 128, ks * 32);
#pragma unroll
    for (int ni = 0; ni < 2; ++ni)
      o[ni] = wmma_bf16(ap, frag_b(Vt, ni * 16, 128, ks * 32), o[ni]);
  }
#pragma unroll
  for (int ni = 0; ni < 2; ++ni) {
#pragma unroll
    for (int r = 0; r < 8; ++r) {
      int s = wave * 16 + r + half * 8;
      int f = ni * 16 + nn;
      if (f < kF)
        Ob[(size_t)(b * kS + s) * kHF + h * kF + f] = f2bf(o[ni][r]);
    }
  }
}

// ---------------- residual add + LayerNorm (fp32 stream + bf16 copy) ----------------
__global__ __launch_bounds__(128) void add_ln_kernel(
    float* __restrict__ x, const float* __restrict__ delta,
    const float* __restrict__ g, const float* __restrict__ beta,
    bf16_t* __restrict__ xb) {
  __shared__ float red[128];
  const int row = blockIdx.x;
  const int tid = threadIdx.x;
  const float* xr = x + (size_t)row * kD;
  const float* dr = delta + (size_t)row * kD;

  float v[3];
#pragma unroll
  for (int i = 0; i < 3; ++i) v[i] = xr[tid + i * 128] + dr[tid + i * 128];

  float s = v[0] + v[1] + v[2];
  red[tid] = s;
  __syncthreads();
  for (int off = 64; off > 0; off >>= 1) {
    if (tid < off) red[tid] += red[tid + off];
    __syncthreads();
  }
  float mean = red[0] * (1.0f / kD);
  __syncthreads();

  float d0 = v[0] - mean, d1 = v[1] - mean, d2 = v[2] - mean;
  red[tid] = d0 * d0 + d1 * d1 + d2 * d2;
  __syncthreads();
  for (int off = 64; off > 0; off >>= 1) {
    if (tid < off) red[tid] += red[tid + off];
    __syncthreads();
  }
  float inv = rsqrtf(red[0] * (1.0f / kD) + 1e-6f);

  float* xw = x + (size_t)row * kD;
  bf16_t* xbw = xb + (size_t)row * kD;
#pragma unroll
  for (int i = 0; i < 3; ++i) {
    int d = tid + i * 128;
    float y = (v[i] - mean) * inv * g[d] + beta[d];
    xw[d] = y;
    xbw[d] = f2bf(y);
  }
}

// ---------------- slice x[:, :L, :] to output ----------------
__global__ __launch_bounds__(128) void output_kernel(const float* __restrict__ x,
                                                     float* __restrict__ out) {
  int row = blockIdx.x;            // b*L + l
  int b = row / kL, l = row - b * kL;
  const float* src = x + (size_t)(b * kS + l) * kD;
  float* dst = out + (size_t)row * kD;
#pragma unroll
  for (int i = 0; i < 3; ++i) dst[threadIdx.x + i * 128] = src[threadIdx.x + i * 128];
}

// ---------------- host orchestration ----------------
extern "C" void kernel_launch(void* const* d_in, const int* in_sizes, int n_in,
                              void* d_out, int out_size, void* d_ws, size_t ws_size,
                              hipStream_t stream) {
  (void)in_sizes; (void)n_in; (void)out_size; (void)ws_size;

  const float* enc      = (const float*)d_in[0];   // [B,S,D]
  const int*   phrase   = (const int*)d_in[1];     // [B,L]
  const float* pos_emb  = (const float*)d_in[2];   // [S,D]
  const float* char_emb = (const float*)d_in[3];   // [NCHAR,D]
  const float* pmha_wq  = (const float*)d_in[4];   // [H,D,F]
  const float* pmha_wk  = (const float*)d_in[5];
  const float* pmha_wv  = (const float*)d_in[6];
  const float* pmha_wo  = (const float*)d_in[7];   // [HF,D]
  const float* pln_g    = (const float*)d_in[8];
  const float* pln_b    = (const float*)d_in[9];
  const float* blk_wq   = (const float*)d_in[10];  // [NB,H,D,F]
  const float* blk_wk   = (const float*)d_in[11];
  const float* blk_wv   = (const float*)d_in[12];
  const float* blk_wo   = (const float*)d_in[13];  // [NB,HF,D]
  const float* ln1_g    = (const float*)d_in[14];  // [NB,D]
  const float* ln1_b    = (const float*)d_in[15];
  const float* ln2_g    = (const float*)d_in[16];
  const float* ln2_b    = (const float*)d_in[17];
  const float* mlp_w1   = (const float*)d_in[18];  // [NB,D,MLP]
  const float* mlp_w2   = (const float*)d_in[19];  // [NB,MLP,D]

  // workspace carve (deterministic)
  char* w = (char*)d_ws;
  auto carve = [&](size_t n) -> char* {
    char* p = w;
    w += (n + 255) & ~(size_t)255;
    return p;
  };
  float*  x_f32  = (float*)carve((size_t)kBS * kD * 4);
  bf16_t* x_bf   = (bf16_t*)carve((size_t)kBS * kD * 2);
  bf16_t* enc_bf = (bf16_t*)carve((size_t)kBS * kD * 2);
  bf16_t* qb     = (bf16_t*)carve((size_t)kBS * kHF * 2);
  bf16_t* kb     = (bf16_t*)carve((size_t)kBS * kHF * 2);
  bf16_t* vb     = (bf16_t*)carve((size_t)kBS * kHF * 2);
  bf16_t* ab     = (bf16_t*)carve((size_t)kBS * kHF * 2);
  float*  tmp    = (float*)carve((size_t)kBS * kD * 4);
  bf16_t* hb     = (bf16_t*)carve((size_t)kBS * kMLP * 2);
  bf16_t* wqT    = (bf16_t*)carve((size_t)kHF * kD * 2);
  bf16_t* wkT    = (bf16_t*)carve((size_t)kHF * kD * 2);
  bf16_t* wvT    = (bf16_t*)carve((size_t)kHF * kD * 2);
  bf16_t* woT    = (bf16_t*)carve((size_t)kD * kHF * 2);
  bf16_t* m1T    = (bf16_t*)carve((size_t)kD * kMLP * 2);
  bf16_t* m2T    = (bf16_t*)carve((size_t)kMLP * kD * 2);

  auto cvtT = [&](const float* in, bf16_t* out, int batch, int R, int C) {
    dim3 g((R * C + 255) / 256, batch);
    transpose_cvt_kernel<<<g, 256, 0, stream>>>(in, out, R, C);
  };
  auto gemm = [&](const bf16_t* A, const bf16_t* Bt, void* C, int M, int N, int K,
                  int mode) {
    dim3 g(N / 64, M / 128);
    gemm_bf16_kernel<<<g, 128, 0, stream>>>(A, Bt, C, M, N, K, mode);
  };
  auto attn_block = [&](const float* wq, const float* wk, const float* wv,
                        const float* wo, const bf16_t* kvsrc,
                        const float* g_, const float* b_) {
    cvtT(wq, wqT, kH, kD, kF);               // -> [HF, D] Bt layout
    cvtT(wk, wkT, kH, kD, kF);
    cvtT(wv, wvT, kH, kD, kF);
    cvtT(wo, woT, 1, kHF, kD);               // -> [D, HF] Bt layout
    gemm(x_bf,  wqT, qb, kBS, kHF, kD, 1);   // Q (bf16)
    gemm(kvsrc, wkT, kb, kBS, kHF, kD, 1);   // K (bf16)
    gemm(kvsrc, wvT, vb, kBS, kHF, kD, 1);   // V (bf16)
    attn_kernel<<<dim3(kH, kB), 256, 0, stream>>>(qb, kb, vb, ab);
    gemm(ab, woT, tmp, kBS, kD, kHF, 0);     // output projection (fp32 delta)
    add_ln_kernel<<<kBS, 128, 0, stream>>>(x_f32, tmp, g_, b_, x_bf);
  };

  // embeddings + encoder convert
  embed_kernel<<<kBS, 128, 0, stream>>>(phrase, pos_emb, char_emb, x_f32, x_bf);
  cvt_kernel<<<(kBS * kD + 255) / 256, 256, 0, stream>>>(enc, enc_bf, kBS * kD);

  // pre-MHA (self-attention, kv = x)
  attn_block(pmha_wq, pmha_wk, pmha_wv, pmha_wo, x_bf, pln_g, pln_b);

  // decoder blocks (cross-attention, kv = encoder outputs) + MLP
  for (int i = 0; i < kNB; ++i) {
    attn_block(blk_wq + (size_t)i * kH * kD * kF,
               blk_wk + (size_t)i * kH * kD * kF,
               blk_wv + (size_t)i * kH * kD * kF,
               blk_wo + (size_t)i * kHF * kD, enc_bf,
               ln1_g + (size_t)i * kD, ln1_b + (size_t)i * kD);

    cvtT(mlp_w1 + (size_t)i * kD * kMLP, m1T, 1, kD, kMLP);   // -> [MLP, D]
    cvtT(mlp_w2 + (size_t)i * kMLP * kD, m2T, 1, kMLP, kD);   // -> [D, MLP]
    gemm(x_bf, m1T, hb, kBS, kMLP, kD, 2);                    // GELU fused, bf16
    gemm(hb, m2T, tmp, kBS, kD, kMLP, 0);                     // fp32 delta
    add_ln_kernel<<<kBS, 128, 0, stream>>>(x_f32, tmp,
                                           ln2_g + (size_t)i * kD,
                                           ln2_b + (size_t)i * kD, x_bf);
  }

  output_kernel<<<kB * kL, 128, 0, stream>>>(x_f32, (float*)d_out);
}
